// RelTM_25391846654699
// MI455X (gfx1250) — matrix operand order repaired and checked by reading, not compiled, not run on hardware
//
#include <hip/hip_runtime.h>

// ---------------------------------------------------------------------------
// Types for CDNA5 WMMA
// ---------------------------------------------------------------------------
typedef __attribute__((ext_vector_type(16))) __bf16 v16bf;
typedef __attribute__((ext_vector_type(8)))  float  v8f;

#define HD 256
#define NH 8
#define DH 32
#define SELF_REL 4

// ---------------------------------------------------------------------------
// helpers
// ---------------------------------------------------------------------------
__device__ __forceinline__ void atomAddF(float* p, float v) {
  __hip_atomic_fetch_add(p, v, __ATOMIC_RELAXED, __HIP_MEMORY_SCOPE_AGENT);
}
// monotonic float->uint key for atomic segment-max
__device__ __forceinline__ unsigned fkey(float x) {
  unsigned u = __float_as_uint(x);
  return (u & 0x80000000u) ? ~u : (u | 0x80000000u);
}
__device__ __forceinline__ float funkey(unsigned k) {
  unsigned u = (k & 0x80000000u) ? (k & 0x7fffffffu) : ~k;
  return __uint_as_float(u);
}
// 32-byte fragment load (two b128)
__device__ __forceinline__ void ldfrag(v16bf& v, const __bf16* p) {
  ((uint4*)&v)[0] = *(const uint4*)(p);
  ((uint4*)&v)[1] = *(const uint4*)(p + 8);
}

// ---------------------------------------------------------------------------
// weight convert + transpose: W[K][Nout] f32  ->  Wt[Nout][K] bf16
// ---------------------------------------------------------------------------
__global__ void k_transpose_cvt(const float* __restrict__ W,
                                __bf16* __restrict__ Wt, int K, int Nout) {
  int idx = blockIdx.x * blockDim.x + threadIdx.x;
  if (idx >= K * Nout) return;
  int k = idx / Nout, n = idx - k * Nout;
  Wt[(size_t)n * K + k] = (__bf16)W[idx];
}

// rel projection: out[10][256] = rel[10][256] @ W[256][256]  (f32, tiny)
__global__ void k_relproj(const float* __restrict__ rel,
                          const float* __restrict__ W,
                          float* __restrict__ out) {
  int idx = blockIdx.x * blockDim.x + threadIdx.x;
  if (idx >= 10 * HD) return;
  int i = idx >> 8, j = idx & 255;
  float s = 0.f;
  for (int k = 0; k < HD; ++k) s += rel[i * HD + k] * W[k * HD + j];
  out[idx] = s;
}

// node embedding gather -> f32 + bf16
__global__ void k_embed(const int* __restrict__ hidx,
                        const float* __restrict__ emb,
                        float* __restrict__ x, __bf16* __restrict__ xb, int N) {
  int idx = blockIdx.x * blockDim.x + threadIdx.x;
  if (idx >= N * HD) return;
  int n = idx >> 8, c = idx & 255;
  float v = emb[hidx[n] * HD + c];
  x[idx] = v;
  xb[idx] = (__bf16)v;
}

// ---------------------------------------------------------------------------
// bf16 WMMA GEMM:  C[M][Nout] = A[M][K] @ Wt[Nout][K]^T
// block = 256 threads (8 waves) -> 64x64 block tile
//   wave tile = 16x32 (two 16x16 f32 accumulators sharing one A fragment)
//   K loop software-pipelined: next A/B fragments prefetched before WMMAs
// flags: 1 = relu, 2 = add bias, 4 = store bf16 (else f32)
// ---------------------------------------------------------------------------
__global__ __launch_bounds__(256)
void k_gemm_wmma(const __bf16* __restrict__ A, const __bf16* __restrict__ Wt,
                 const float* __restrict__ bias, float* __restrict__ Cf,
                 __bf16* __restrict__ Cb, int M, int K, int Nout, int flags) {
  __shared__ __align__(16) __bf16 sW[64 * 256];   // 32 KB weight slab

  const int tid  = threadIdx.x;
  const int wave = tid >> 5;
  const int lane = tid & 31;
  const int h    = lane >> 4;      // lane half
  const int ln   = lane & 15;
  const int rt   = wave & 3;       // row tile 0..3   (16 rows each)
  const int cg   = wave >> 2;      // col group 0..1  (32 cols each)
  const int rowbase = blockIdx.x * 64 + rt * 16;
  const int colbase = blockIdx.y * 64;
  const int cgbase  = colbase + cg * 32;

  v8f acc0 = {}, acc1 = {};

  const int nchunks = K >> 8;
  for (int kc = 0; kc < nchunks; ++kc) {
    // stage Wt[colbase..colbase+63][kc*256 .. +255] into LDS (8-elt vectors)
    for (int v = tid; v < 2048; v += 256) {
      int c  = v >> 5;
      int k8 = v & 31;
      ((uint4*)sW)[c * 32 + k8] =
          *(const uint4*)(Wt + (size_t)(colbase + c) * K + kc * 256 + k8 * 8);
    }
    __syncthreads();

    const __bf16* Abase = A + (size_t)(rowbase + ln) * K + kc * 256;
    const __bf16* B0 = sW + (cg * 32 + ln) * 256;        // cols cgbase..+15
    const __bf16* B1 = B0 + 16 * 256;                    // cols cgbase+16..+31

    v16bf a = {}, b0 = {}, b1 = {}, an = {}, b0n = {}, b1n = {};
    // A 16x32 fragment: lane half h holds K = 16h .. 16h+15 at its row
    ldfrag(a,  Abase + 16 * h);
    // B 32x16 fragment: lane half h holds K = 16h .. 16h+15 at its column
    ldfrag(b0, B0 + 16 * h);
    ldfrag(b1, B1 + 16 * h);
#pragma unroll
    for (int ks = 0; ks < 256; ks += 32) {
      if (ks < 224) {  // prefetch next step while WMMAs run
        ldfrag(an,  Abase + ks + 32 + 16 * h);
        ldfrag(b0n, B0 + ks + 32 + 16 * h);
        ldfrag(b1n, B1 + ks + 32 + 16 * h);
      }
      acc0 = __builtin_amdgcn_wmma_f32_16x16x32_bf16(
          false, a, false, b0, (short)0, acc0, false, false);
      acc1 = __builtin_amdgcn_wmma_f32_16x16x32_bf16(
          false, a, false, b1, (short)0, acc1, false, false);
      a = an; b0 = b0n; b1 = b1n;
    }
    __syncthreads();
  }

  // epilogue: C layout per tile: VGPR i -> row = i + 8h, col = ln
  float bv0 = (flags & 2) ? bias[cgbase + ln]      : 0.f;
  float bv1 = (flags & 2) ? bias[cgbase + 16 + ln] : 0.f;
#pragma unroll
  for (int i = 0; i < 8; ++i) {
    int r = rowbase + i + 8 * h;
    float v0 = acc0[i] + bv0;
    float v1 = acc1[i] + bv1;
    if (flags & 1) { v0 = fmaxf(v0, 0.f); v1 = fmaxf(v1, 0.f); }
    if (flags & 4) {
      Cb[(size_t)r * Nout + cgbase + ln]      = (__bf16)v0;
      Cb[(size_t)r * Nout + cgbase + 16 + ln] = (__bf16)v1;
    } else {
      Cf[(size_t)r * Nout + cgbase + ln]      = v0;
      Cf[(size_t)r * Nout + cgbase + 16 + ln] = v1;
    }
  }
}

// ---------------------------------------------------------------------------
// attention edge kernels  (Etot = E + N, tail edges are self-loops rel=4)
// ---------------------------------------------------------------------------
__global__ void k_edge_scores(const int* __restrict__ esrc, const int* __restrict__ edst,
                              const int* __restrict__ erel,
                              const __bf16* __restrict__ qb, const __bf16* __restrict__ kb,
                              const float* __restrict__ relkp,
                              float* __restrict__ s, unsigned* __restrict__ mkey,
                              int E, int N) {
  int idx = blockIdx.x * blockDim.x + threadIdx.x;
  int tot = (E + N) * NH;
  if (idx >= tot) return;
  int e = idx >> 3, hh = idx & 7;
  int src, dst, rel;
  if (e < E) { src = esrc[e]; dst = edst[e]; rel = erel[e]; }
  else       { src = dst = e - E; rel = SELF_REL; }
  const __bf16* qp = qb + (size_t)dst * HD + hh * DH;
  const __bf16* kp = kb + (size_t)src * HD + hh * DH;
  const float*  rp = relkp + rel * HD + hh * DH;
  float acc = 0.f;
#pragma unroll 8
  for (int d = 0; d < DH; ++d) acc += (float)qp[d] * ((float)kp[d] + rp[d]);
  acc *= 0.1767766953f;   // 1/sqrt(32)
  s[idx] = acc;
  atomicMax(&mkey[dst * NH + hh], fkey(acc));
}

__global__ void k_edge_exp(const int* __restrict__ edst,
                           float* __restrict__ s, const unsigned* __restrict__ mkey,
                           float* __restrict__ den, int E, int N) {
  int idx = blockIdx.x * blockDim.x + threadIdx.x;
  int tot = (E + N) * NH;
  if (idx >= tot) return;
  int e = idx >> 3, hh = idx & 7;
  int dst = (e < E) ? edst[e] : (e - E);
  float ex = __expf(s[idx] - funkey(mkey[dst * NH + hh]));
  s[idx] = ex;
  atomAddF(&den[dst * NH + hh], ex);
}

__global__ void k_edge_accum(const int* __restrict__ esrc, const int* __restrict__ edst,
                             const int* __restrict__ erel,
                             const float* __restrict__ s, const float* __restrict__ den,
                             const __bf16* __restrict__ vb, const float* __restrict__ relvp,
                             float* __restrict__ att, int E, int N) {
  int idx = blockIdx.x * blockDim.x + threadIdx.x;
  int tot = (E + N) * NH;
  if (idx >= tot) return;
  int e = idx >> 3, hh = idx & 7;
  int src, dst, rel;
  if (e < E) { src = esrc[e]; dst = edst[e]; rel = erel[e]; }
  else       { src = dst = e - E; rel = SELF_REL; }
  float a = s[idx] / den[dst * NH + hh];
  const __bf16* vp = vb + (size_t)src * HD + hh * DH;
  const float*  rp = relvp + rel * HD + hh * DH;
  float* ap = att + (size_t)dst * HD + hh * DH;
#pragma unroll 8
  for (int d = 0; d < DH; ++d) atomAddF(&ap[d], a * ((float)vp[d] + rp[d]));
}

__global__ void k_cvt_bf16(const float* __restrict__ in, __bf16* __restrict__ out, int n) {
  int idx = blockIdx.x * blockDim.x + threadIdx.x;
  if (idx < n) out[idx] = (__bf16)in[idx];
}

// ---------------------------------------------------------------------------
// residual + layernorm (one wave32 per node), writes f32 + bf16
// ---------------------------------------------------------------------------
__global__ __launch_bounds__(256)
void k_lnorm(const float* __restrict__ base, const float* __restrict__ delta,
             const float* __restrict__ g, const float* __restrict__ b,
             float* __restrict__ outf, __bf16* __restrict__ outb, int N) {
  int node = blockIdx.x * 8 + (threadIdx.x >> 5);
  int lane = threadIdx.x & 31;
  if (node >= N) return;
  const float* xp = base  + (size_t)node * HD;
  const float* dp = delta + (size_t)node * HD;
  float v[8], s = 0.f, s2 = 0.f;
#pragma unroll
  for (int j = 0; j < 8; ++j) {
    float t = xp[lane + 32 * j] + dp[lane + 32 * j];
    v[j] = t; s += t; s2 += t * t;
  }
#pragma unroll
  for (int off = 16; off; off >>= 1) {
    s  += __shfl_xor(s,  off, 32);
    s2 += __shfl_xor(s2, off, 32);
  }
  float mu  = s * (1.f / 256.f);
  float var = s2 * (1.f / 256.f) - mu * mu;
  float inv = rsqrtf(var + 1e-5f);
#pragma unroll
  for (int j = 0; j < 8; ++j) {
    int c = lane + 32 * j;
    float y = (v[j] - mu) * inv * g[c] + b[c];
    outf[(size_t)node * HD + c] = y;
    outb[(size_t)node * HD + c] = (__bf16)y;
  }
}

// ---------------------------------------------------------------------------
// graph mean-pool + readout MLP (256->128->64->1)
// ---------------------------------------------------------------------------
__global__ void k_pool(const float* __restrict__ x, const int* __restrict__ gid,
                       float* __restrict__ hg, float* __restrict__ cnt, int N) {
  int idx = blockIdx.x * blockDim.x + threadIdx.x;
  if (idx >= N * HD) return;
  int n = idx >> 8, c = idx & 255;
  int gg = gid[n];
  atomAddF(&hg[(size_t)gg * HD + c], x[idx]);
  if (c == 0) atomAddF(&cnt[gg], 1.f);
}

__global__ __launch_bounds__(256)
void k_readout(const float* __restrict__ hg, const float* __restrict__ cnt,
               const float* __restrict__ w0, const float* __restrict__ b0,
               const float* __restrict__ w1, const float* __restrict__ b1,
               const float* __restrict__ w2, const float* __restrict__ b2,
               float* __restrict__ out) {
  __shared__ float sx[256], y1[128], y2[64];
  int gg = blockIdx.x, t = threadIdx.x;
  float c = fmaxf(cnt[gg], 1.f);
  sx[t] = hg[(size_t)gg * HD + t] / c;
  __syncthreads();
  if (t < 128) {
    float s = b0[t];
    for (int k = 0; k < 256; ++k) s += sx[k] * w0[k * 128 + t];
    y1[t] = fmaxf(s, 0.f);
  }
  __syncthreads();
  if (t < 64) {
    float s = b1[t];
    for (int k = 0; k < 128; ++k) s += y1[k] * w1[k * 64 + t];
    y2[t] = fmaxf(s, 0.f);
  }
  __syncthreads();
  if (t == 0) {
    float s = b2[0];
    for (int k = 0; k < 64; ++k) s += y2[k] * w2[k];
    out[gg] = s;
  }
}

// ---------------------------------------------------------------------------
// host
// ---------------------------------------------------------------------------
extern "C" void kernel_launch(void* const* d_in, const int* in_sizes, int n_in,
                              void* d_out, int out_size, void* d_ws, size_t ws_size,
                              hipStream_t stream) {
  const int N = in_sizes[0];
  const int E = in_sizes[1];
  const int G = out_size;
  const int Etot = E + N;

  const int*   hix   = (const int*)d_in[0];
  const int*   esrc  = (const int*)d_in[1];
  const int*   edst  = (const int*)d_in[2];
  const int*   erel  = (const int*)d_in[3];
  const int*   gid   = (const int*)d_in[4];
  const float* emb   = (const float*)d_in[5];
  const float* Wq    = (const float*)d_in[6];
  const float* Wk    = (const float*)d_in[7];
  const float* Wv    = (const float*)d_in[8];
  const float* Wo    = (const float*)d_in[9];
  const float* relk  = (const float*)d_in[10];
  const float* relv  = (const float*)d_in[11];
  const float* ln1g  = (const float*)d_in[12];
  const float* ln1b  = (const float*)d_in[13];
  const float* W1    = (const float*)d_in[14];
  const float* B1    = (const float*)d_in[15];
  const float* W2    = (const float*)d_in[16];
  const float* B2    = (const float*)d_in[17];
  const float* ln2g  = (const float*)d_in[18];
  const float* ln2b  = (const float*)d_in[19];
  const float* r0w   = (const float*)d_in[20];
  const float* r0b   = (const float*)d_in[21];
  const float* r1w   = (const float*)d_in[22];
  const float* r1b   = (const float*)d_in[23];
  const float* r2w   = (const float*)d_in[24];
  const float* r2b   = (const float*)d_in[25];
  float* out = (float*)d_out;

  // ---- workspace layout (aliased) ----
  char* ws = (char*)d_ws;
  size_t off = 0;
  auto give = [&](size_t bytes) -> size_t {
    size_t o = off; off += (bytes + 255) & ~(size_t)255; return o;
  };
  size_t o_x    = give((size_t)N * HD * 4);
  size_t o_xb   = give((size_t)N * HD * 2);   // also tb after LN1
  size_t o_qb   = give((size_t)N * HD * 2);   // q/k/v/attb slab = FFN hidden later
  size_t o_kb   = give((size_t)N * HD * 2);
  size_t o_vb   = give((size_t)N * HD * 2);
  size_t o_attb = give((size_t)N * HD * 2);
  size_t o_s    = give((size_t)Etot * NH * 4);
  size_t o_mk   = give((size_t)N * NH * 4);
  size_t o_den  = give((size_t)N * NH * 4);
  size_t o_att  = give((size_t)N * HD * 4);   // also t after attention consumed
  size_t o_o    = give((size_t)N * HD * 4);   // also f2 after LN1 consumed
  size_t o_wt   = give((size_t)(8 * 65536 + 4 * 262144) * 2);
  size_t o_rk   = give((size_t)2 * 10 * HD * 4);
  size_t o_rv   = give((size_t)2 * 10 * HD * 4);
  size_t o_hg   = give((size_t)G * HD * 4);
  size_t o_cnt  = give((size_t)G * 4);
  if (off > ws_size) return;  // workspace too small: bail safely

  float*  xf   = (float*)(ws + o_x);
  __bf16* xb   = (__bf16*)(ws + o_xb);
  __bf16* qb   = (__bf16*)(ws + o_qb);
  __bf16* kb   = (__bf16*)(ws + o_kb);
  __bf16* vb   = (__bf16*)(ws + o_vb);
  __bf16* attb = (__bf16*)(ws + o_attb);
  __bf16* hid  = qb;                       // N x 1024 bf16 reuses q/k/v/attb slab
  float*  sbuf = (float*)(ws + o_s);
  unsigned* mk = (unsigned*)(ws + o_mk);
  float*  den  = (float*)(ws + o_den);
  float*  att  = (float*)(ws + o_att);
  float*  tbuf = att;                      // t reuses att
  __bf16* tb   = xb;                       // tb reuses xb
  float*  obuf = (float*)(ws + o_o);
  float*  f2   = obuf;                     // f2 reuses o
  __bf16* wta  = (__bf16*)(ws + o_wt);
  float*  rkp  = (float*)(ws + o_rk);
  float*  rvp  = (float*)(ws + o_rv);
  float*  hg   = (float*)(ws + o_hg);
  float*  cnt  = (float*)(ws + o_cnt);

  __bf16* wqT[2], *wkT[2], *wvT[2], *woT[2], *w1T[2], *w2T[2];
  for (int l = 0; l < 2; ++l) {
    wqT[l] = wta + (size_t)l * 65536;
    wkT[l] = wta + (size_t)(2 + l) * 65536;
    wvT[l] = wta + (size_t)(4 + l) * 65536;
    woT[l] = wta + (size_t)(6 + l) * 65536;
    w1T[l] = wta + (size_t)8 * 65536 + (size_t)l * 262144;
    w2T[l] = wta + (size_t)8 * 65536 + (size_t)(2 + l) * 262144;
  }

  const int T = 256;
  auto cdiv = [](long long a, long long b) { return (int)((a + b - 1) / b); };

  // ---- one-time per call: weights -> bf16 transposed, rel projections, embed
  for (int l = 0; l < 2; ++l) {
    k_transpose_cvt<<<cdiv(65536, T),  T, 0, stream>>>(Wq + (size_t)l * 65536,  wqT[l], 256, 256);
    k_transpose_cvt<<<cdiv(65536, T),  T, 0, stream>>>(Wk + (size_t)l * 65536,  wkT[l], 256, 256);
    k_transpose_cvt<<<cdiv(65536, T),  T, 0, stream>>>(Wv + (size_t)l * 65536,  wvT[l], 256, 256);
    k_transpose_cvt<<<cdiv(65536, T),  T, 0, stream>>>(Wo + (size_t)l * 65536,  woT[l], 256, 256);
    k_transpose_cvt<<<cdiv(262144, T), T, 0, stream>>>(W1 + (size_t)l * 262144, w1T[l], 256, 1024);
    k_transpose_cvt<<<cdiv(262144, T), T, 0, stream>>>(W2 + (size_t)l * 262144, w2T[l], 1024, 256);
    k_relproj<<<cdiv(2560, T), T, 0, stream>>>(relk + (size_t)l * 2560, Wk + (size_t)l * 65536, rkp + (size_t)l * 2560);
    k_relproj<<<cdiv(2560, T), T, 0, stream>>>(relv + (size_t)l * 2560, Wv + (size_t)l * 65536, rvp + (size_t)l * 2560);
  }
  k_embed<<<cdiv((long long)N * HD, T), T, 0, stream>>>(hix, emb, xf, xb, N);

  dim3 g256(N / 64, 256 / 64);     // Nout = 256
  dim3 g1024(N / 64, 1024 / 64);   // Nout = 1024
  int egrid = cdiv((long long)Etot * NH, T);

  for (int l = 0; l < 2; ++l) {
    for (int rep = 0; rep < 4; ++rep) {
      // Q/K/V projections (bf16 out)
      k_gemm_wmma<<<g256, T, 0, stream>>>(xb, wqT[l], nullptr, nullptr, qb, N, 256, 256, 4);
      k_gemm_wmma<<<g256, T, 0, stream>>>(xb, wkT[l], nullptr, nullptr, kb, N, 256, 256, 4);
      k_gemm_wmma<<<g256, T, 0, stream>>>(xb, wvT[l], nullptr, nullptr, vb, N, 256, 256, 4);
      // attention
      hipMemsetAsync(mk,  0, (size_t)N * NH * 4, stream);
      hipMemsetAsync(den, 0, (size_t)N * NH * 4, stream);
      hipMemsetAsync(att, 0, (size_t)N * HD * 4, stream);
      k_edge_scores<<<egrid, T, 0, stream>>>(esrc, edst, erel, qb, kb,
                                             rkp + (size_t)l * 2560, sbuf, mk, E, N);
      k_edge_exp<<<egrid, T, 0, stream>>>(edst, sbuf, mk, den, E, N);
      k_edge_accum<<<egrid, T, 0, stream>>>(esrc, edst, erel, sbuf, den, vb,
                                            rvp + (size_t)l * 2560, att, E, N);
      k_cvt_bf16<<<cdiv((long long)N * HD, T), T, 0, stream>>>(att, attb, N * HD);
      // output projection + residual LN1
      k_gemm_wmma<<<g256, T, 0, stream>>>(attb, woT[l], nullptr, obuf, nullptr, N, 256, 256, 0);
      k_lnorm<<<N / 8, T, 0, stream>>>(xf, obuf, ln1g + (size_t)l * HD, ln1b + (size_t)l * HD,
                                       tbuf, tb, N);
      // FFN (relu+bias fused, bf16 hidden) + residual LN2 -> new x
      k_gemm_wmma<<<g1024, T, 0, stream>>>(tb, w1T[l], B1 + (size_t)l * 1024, nullptr, hid,
                                           N, 256, 1024, 1 | 2 | 4);
      k_gemm_wmma<<<g256, T, 0, stream>>>(hid, w2T[l], B2 + (size_t)l * 256, f2, nullptr,
                                          N, 1024, 256, 2);
      k_lnorm<<<N / 8, T, 0, stream>>>(tbuf, f2, ln2g + (size_t)l * HD, ln2b + (size_t)l * HD,
                                       xf, xb, N);
    }
  }

  // ---- readout ----
  hipMemsetAsync(hg,  0, (size_t)G * HD * 4, stream);
  hipMemsetAsync(cnt, 0, (size_t)G * 4, stream);
  k_pool<<<cdiv((long long)N * HD, T), T, 0, stream>>>(xf, gid, hg, cnt, N);
  k_readout<<<G, T, 0, stream>>>(hg, cnt, r0w, r0b, r1w, r1b, r2w, r2b, out);
}
